// ResidualGATBlock_22978075033938
// MI455X (gfx1250) — matrix operand order repaired and checked by reading, not compiled, not run on hardware
//
#include <hip/hip_runtime.h>
#include <hip/hip_bf16.h>
#include <math.h>

typedef __attribute__((ext_vector_type(16))) _Float16 v16h;
typedef __attribute__((ext_vector_type(8)))  _Float16 v8h;
typedef __attribute__((ext_vector_type(8)))  float    v8f;
typedef __attribute__((ext_vector_type(4)))  float    v4f;
typedef __attribute__((ext_vector_type(4)))  int      v4i;

#define DD 128
#define HH 4
#define NEG_SLOPE 0.2f
#define LN_EPS 1e-5f

// Padded WMMA-B fragment: 512 payload halves + per-lane pad for LDS banking.
// Lane's 16 halves live at  frag*FRAG_H + lane*16 + (lane>>3)*8.
#define FRAG_H 544

#if __has_builtin(__builtin_amdgcn_global_load_async_to_lds_b128)
#define HAVE_ASYNC_LDS 1
#else
#define HAVE_ASYNC_LDS 0
#endif

__device__ __forceinline__ int lane_hoff(int lane) {
    return lane * 16 + (lane >> 3) * 8;
}

__device__ __forceinline__ float gelu_erf(float v) {
    return 0.5f * v * (1.0f + erff(v * 0.70710678118654752f));
}

__device__ __forceinline__ float wave_sum(float v) {
#pragma unroll
    for (int off = 16; off > 0; off >>= 1) v += __shfl_xor(v, off, 32);
    return v;
}

// Cooperative block copy global -> LDS in 16B chunks (async-to-LDS when available).
__device__ __forceinline__ void stage_to_lds(const _Float16* __restrict__ gsrc,
                                             _Float16* lds, int bytes) {
    for (int off = threadIdx.x * 16; off < bytes; off += 256 * 16) {
#if HAVE_ASYNC_LDS
        __builtin_amdgcn_global_load_async_to_lds_b128(
            (__attribute__((address_space(1))) v4i*)((char*)gsrc + off),
            (__attribute__((address_space(3))) v4i*)((char*)lds + off), 0, 0);
#else
        *(v8h*)((char*)lds + off) = *(const v8h*)((const char*)gsrc + off);
#endif
    }
#if HAVE_ASYNC_LDS
#if __has_builtin(__builtin_amdgcn_s_wait_asynccnt)
    __builtin_amdgcn_s_wait_asynccnt(0);
#else
    asm volatile("s_wait_asynccnt 0" ::: "memory");
#endif
#endif
    __syncthreads();
}

// ---------------------------------------------------------------------------
// Convert a row-major f32 weight [K, Ncols] into padded f16 WMMA-B fragments.
// Fragment (kt, nt): element (lane, i) = W[kt*32 + (lane>>4)*8 + (i&7) + (i>>3)*16][nt*16 + (lane&15)]
// stored at frag*FRAG_H + lane_hoff(lane) + i.
// ---------------------------------------------------------------------------
__global__ void prep_weight_swz(const float* __restrict__ W, _Float16* __restrict__ Wz,
                                int K, int Ncols) {
    int idx = blockIdx.x * blockDim.x + threadIdx.x;
    int total = K * Ncols;
    if (idx >= total) return;
    int frag = idx >> 9;
    int r    = idx & 511;
    int lane = r >> 4;
    int i    = r & 15;
    int ntiles = Ncols >> 4;
    int kt = frag / ntiles;
    int nt = frag - kt * ntiles;
    int k = kt * 32 + (lane >> 4) * 8 + (i & 7) + (i >> 3) * 16;
    int n = nt * 16 + (lane & 15);
    Wz[(size_t)frag * FRAG_H + lane_hoff(lane) + i] = (_Float16)W[(size_t)k * Ncols + n];
}

// time_cond = gelu(time_emb @ tW + tb)   [B, 128]
__global__ void time_cond_kernel(const float* __restrict__ te, const float* __restrict__ tW,
                                 const float* __restrict__ tb, float* __restrict__ tc, int Bn) {
    int idx = blockIdx.x * blockDim.x + threadIdx.x;
    if (idx >= Bn * DD) return;
    int b = idx / DD, d = idx - b * DD;
    float acc = tb[d];
    for (int k = 0; k < 128; ++k) acc += te[b * 128 + k] * tW[k * DD + d];
    tc[idx] = gelu_erf(acc);
}

__global__ void init_softmax_kernel(float* __restrict__ nmax, float* __restrict__ den, int n) {
    int i = blockIdx.x * blockDim.x + threadIdx.x;
    if (i < n) { nmax[i] = -__builtin_inff(); den[i] = 0.0f; }
}

__global__ void zero_kernel(float* __restrict__ p, int n) {
    int i = blockIdx.x * blockDim.x + threadIdx.x;
    if (i < n) p[i] = 0.0f;
}

// LN1: wave per row, f16 output (WMMA A operand source)
__global__ __launch_bounds__(256) void ln1_kernel(const float* __restrict__ x,
        const float* __restrict__ g, const float* __restrict__ b,
        _Float16* __restrict__ h16, int N) {
    int row  = (blockIdx.x * blockDim.x + threadIdx.x) >> 5;
    int lane = threadIdx.x & 31;
    if (row >= N) return;
    v4f v = *(const v4f*)(x + (size_t)row * DD + lane * 4);
    float s = 0.f, ss = 0.f;
#pragma unroll
    for (int i = 0; i < 4; ++i) { s += v[i]; ss += v[i] * v[i]; }
    s = wave_sum(s); ss = wave_sum(ss);
    float mu  = s * (1.0f / 128.0f);
    float var = ss * (1.0f / 128.0f) - mu * mu;
    float rs  = rsqrtf(var + LN_EPS);
    _Float16* o = h16 + (size_t)row * DD + lane * 4;
#pragma unroll
    for (int i = 0; i < 4; ++i) {
        int k = lane * 4 + i;
        o[i] = (_Float16)((v[i] - mu) * rs * g[k] + b[k]);
    }
}

// Load a 16x32 f16 A fragment from a row-major f16 matrix (ld in halves).
// Lane layout: row = lane&15; K chunks [c0, c0+8) and [c0+16, c0+24), c0 = (lane>>4)*8.
__device__ __forceinline__ v16h load_a_frag(const _Float16* __restrict__ A, size_t ld,
                                            int row, int k0, int lane) {
    int c0 = (lane >> 4) * 8;
    const _Float16* pa = A + (size_t)row * ld + k0 + c0;
    union { v16h v; v8h h[2]; } au;
    au.h[0] = *(const v8h*)pa;
    au.h[1] = *(const v8h*)(pa + 16);
    return au.v;
}

// Read one lane's B-fragment halves from LDS (two 16B ds loads, 16B aligned).
__device__ __forceinline__ v16h lds_b_frag(const _Float16* p) {
    union { v16h v; v8h h[2]; } u;
    u.h[0] = *(const v8h*)p;
    u.h[1] = *(const v8h*)(p + 8);
    return u.v;
}

// Dual GEMM: feat_src = h@Wsrc+bsrc, feat_dst = h@Wdst+bdst. 64 rows/block,
// 8 waves: waves 0-3 -> src (row subtile w), waves 4-7 -> dst.
// Weights staged in LDS (async copy); A fragments preloaded in registers.
__global__ __launch_bounds__(256) void gemm_feat_kernel(const _Float16* __restrict__ h16,
        const _Float16* __restrict__ wsrcz, const _Float16* __restrict__ wdstz,
        const float* __restrict__ bsrc, const float* __restrict__ bdst,
        float* __restrict__ fsrc, float* __restrict__ fdst, int N) {
    __shared__ _Float16 smem[2 * 32 * FRAG_H];   // 69,632 B
    int wv = threadIdx.x >> 5, lane = threadIdx.x & 31;
    int mat = wv >> 2, rsub = wv & 3;
    int rowbase = blockIdx.x * 64 + rsub * 16;
    const float* bias = mat ? bdst : bsrc;
    float* out        = mat ? fdst : fsrc;

    stage_to_lds(wsrcz, smem, 32 * FRAG_H * 2);
    stage_to_lds(wdstz, smem + 32 * FRAG_H, 32 * FRAG_H * 2);

    int row_a = rowbase + (lane & 15);
    if (row_a > N - 1) row_a = N - 1;
    v16h a[4];
#pragma unroll
    for (int kt = 0; kt < 4; ++kt) a[kt] = load_a_frag(h16, DD, row_a, kt * 32, lane);

    const _Float16* wl = smem + (size_t)mat * 32 * FRAG_H + lane_hoff(lane);
    v8f acc[8] = {};
#pragma unroll
    for (int f = 0; f < 32; ++f) {            // frag f = kt*8 + nt (linear)
        v16h b = lds_b_frag(wl + (size_t)f * FRAG_H);
        int kt = f >> 3, nt = f & 7;
        acc[nt] = __builtin_amdgcn_wmma_f32_16x16x32_f16(
            false, a[kt], false, b, (short)0, acc[nt], false, false);
    }

    int nlane = lane & 15, mh = (lane >> 4) * 8;
#pragma unroll
    for (int nt = 0; nt < 8; ++nt) {
        int col = nt * 16 + nlane;
        float bc = bias[col];
#pragma unroll
        for (int j = 0; j < 8; ++j) {
            int row = rowbase + mh + j;
            if (row < N) out[(size_t)row * DD + col] = acc[nt][j] + bc;
        }
    }
}

// Wave per edge: gather both endpoint features (16B/lane), leaky-relu, per-head
// dot with attn, 8-lane group reduce, store score + float atomic-max into dst node.
__global__ __launch_bounds__(256) void edge_score_kernel(const int* __restrict__ src,
        const int* __restrict__ dst, const float* __restrict__ fsrc,
        const float* __restrict__ fdst, const float* __restrict__ attn,
        float* __restrict__ scores, float* __restrict__ nmax, int E) {
    int e = (blockIdx.x * blockDim.x + threadIdx.x) >> 5;
    int lane = threadIdx.x & 31;
    if (e >= E) return;
    int s = src[e], d = dst[e];
    v4f a = *(const v4f*)(fsrc + (size_t)s * DD + lane * 4);
    v4f b = *(const v4f*)(fdst + (size_t)d * DD + lane * 4);
    float dot = 0.f;
#pragma unroll
    for (int i = 0; i < 4; ++i) {
        float v = a[i] + b[i];
        v = v > 0.f ? v : NEG_SLOPE * v;
        dot += v * attn[lane * 4 + i];
    }
#pragma unroll
    for (int off = 1; off < 8; off <<= 1) dot += __shfl_xor(dot, off, 32);
    if ((lane & 7) == 0) {
        int h = lane >> 3;
        scores[(size_t)e * HH + h] = dot;
        float* addr = nmax + (size_t)d * HH + h;
        if (dot >= 0.f) atomicMax((int*)addr, __float_as_int(dot));
        else            atomicMin((unsigned int*)addr, (unsigned int)__float_as_int(dot));
    }
}

// ex = exp(score - max[dst]); in-place over scores; atomic-add into denom.
__global__ void edge_exp_kernel(const int* __restrict__ dst, float* __restrict__ scores,
        const float* __restrict__ nmax, float* __restrict__ den, int E) {
    int i = blockIdx.x * blockDim.x + threadIdx.x;
    if (i >= E * HH) return;
    int e = i >> 2, h = i & 3;
    int d = dst[e];
    float ex = expf(scores[i] - nmax[d * HH + h]);
    scores[i] = ex;
    atomicAdd(den + d * HH + h, ex);
}

// Wave per edge: alpha-weighted scatter of feat_src into rst[dst].
__global__ __launch_bounds__(256) void edge_agg_kernel(const int* __restrict__ src,
        const int* __restrict__ dst, const float* __restrict__ fsrc,
        const float* __restrict__ ex, const float* __restrict__ den,
        float* __restrict__ rst, int E) {
    int e = (blockIdx.x * blockDim.x + threadIdx.x) >> 5;
    int lane = threadIdx.x & 31;
    if (e >= E) return;
    int s = src[e], d = dst[e];
    int h = lane >> 3;
    float alpha = ex[(size_t)e * HH + h] / den[(size_t)d * HH + h];
    v4f a = *(const v4f*)(fsrc + (size_t)s * DD + lane * 4);
    float* o = rst + (size_t)d * DD + lane * 4;
#pragma unroll
    for (int i = 0; i < 4; ++i) atomicAdd(o + i, a[i] * alpha);
}

// hres = gelu(rst + gat_bias) + time_cond[gid] + x  (in-place over rst),
// then LN2 -> f16 FFN input. Wave per row.
__global__ __launch_bounds__(256) void node_mid_kernel(const float* __restrict__ x,
        const int* __restrict__ gids, const float* __restrict__ tcond,
        const float* __restrict__ gbias, const float* __restrict__ g2,
        const float* __restrict__ b2, float* __restrict__ rst_hres,
        _Float16* __restrict__ h2in16, int N) {
    int row  = (blockIdx.x * blockDim.x + threadIdx.x) >> 5;
    int lane = threadIdx.x & 31;
    if (row >= N) return;
    int gid = gids[row];
    float* rrow = rst_hres + (size_t)row * DD + lane * 4;
    const float* xrow = x + (size_t)row * DD + lane * 4;
    const float* trow = tcond + (size_t)gid * DD + lane * 4;
    float v[4], s = 0.f, ss = 0.f;
#pragma unroll
    for (int i = 0; i < 4; ++i) {
        int k = lane * 4 + i;
        float t = gelu_erf(rrow[i] + gbias[k]) + trow[i] + xrow[i];
        v[i] = t; s += t; ss += t * t;
    }
#pragma unroll
    for (int i = 0; i < 4; ++i) rrow[i] = v[i];  // hres
    s = wave_sum(s); ss = wave_sum(ss);
    float mu  = s * (1.0f / 128.0f);
    float var = ss * (1.0f / 128.0f) - mu * mu;
    float rs  = rsqrtf(var + LN_EPS);
    _Float16* o = h2in16 + (size_t)row * DD + lane * 4;
#pragma unroll
    for (int i = 0; i < 4; ++i) {
        int k = lane * 4 + i;
        o[i] = (_Float16)((v[i] - mu) * rs * g2[k] + b2[k]);
    }
}

// FFN1: y = gelu(h2in @ W1 + b1)  [N,256] f16.  64 rows/block, 8 waves:
// wave = (colhalf<<2)|rsub; each wave 16 rows x 128 cols (8 tiles).
__global__ __launch_bounds__(256) void ffn1_kernel(const _Float16* __restrict__ h2in16,
        const _Float16* __restrict__ w1z, const float* __restrict__ b1,
        _Float16* __restrict__ y16, int N) {
    __shared__ _Float16 smem[64 * FRAG_H];       // 69,632 B
    int wv = threadIdx.x >> 5, lane = threadIdx.x & 31;
    int rsub = wv & 3, chalf = wv >> 2;
    int rowbase = blockIdx.x * 64 + rsub * 16;

    stage_to_lds(w1z, smem, 64 * FRAG_H * 2);

    int row_a = rowbase + (lane & 15);
    if (row_a > N - 1) row_a = N - 1;
    v16h a[4];
#pragma unroll
    for (int kt = 0; kt < 4; ++kt) a[kt] = load_a_frag(h2in16, DD, row_a, kt * 32, lane);

    // wave's B frags: frag(kt, t) = kt*16 + chalf*8 + t
    const _Float16* wl = smem + (size_t)(chalf * 8) * FRAG_H + lane_hoff(lane);
    v8f acc[8] = {};
#pragma unroll
    for (int f = 0; f < 32; ++f) {
        int kt = f >> 3, t = f & 7;
        v16h b = lds_b_frag(wl + ((size_t)kt * 16 + t) * FRAG_H);
        acc[t] = __builtin_amdgcn_wmma_f32_16x16x32_f16(
            false, a[kt], false, b, (short)0, acc[t], false, false);
    }

    int nlane = lane & 15, mh = (lane >> 4) * 8;
#pragma unroll
    for (int t = 0; t < 8; ++t) {
        int col = (chalf * 8 + t) * 16 + nlane;
        float bc = b1[col];
#pragma unroll
        for (int j = 0; j < 8; ++j) {
            int row = rowbase + mh + j;
            if (row < N)
                y16[(size_t)row * 256 + col] = (_Float16)gelu_erf(acc[t][j] + bc);
        }
    }
}

// FFN2 + final residual: out = hres + (y @ W2 + b2).  K=256 (8 ktiles).
__global__ __launch_bounds__(256) void ffn2_kernel(const _Float16* __restrict__ y16,
        const _Float16* __restrict__ w2z, const float* __restrict__ b2,
        const float* __restrict__ hres, float* __restrict__ out, int N) {
    __shared__ _Float16 smem[64 * FRAG_H];       // 69,632 B
    int wv = threadIdx.x >> 5, lane = threadIdx.x & 31;
    int rsub = wv & 3, chalf = wv >> 2;
    int rowbase = blockIdx.x * 64 + rsub * 16;

    stage_to_lds(w2z, smem, 64 * FRAG_H * 2);

    int row_a = rowbase + (lane & 15);
    if (row_a > N - 1) row_a = N - 1;
    v16h a[8];
#pragma unroll
    for (int kt = 0; kt < 8; ++kt) a[kt] = load_a_frag(y16, 256, row_a, kt * 32, lane);

    // wave's B frags: frag(kt, t) = kt*8 + chalf*4 + t
    const _Float16* wl = smem + (size_t)(chalf * 4) * FRAG_H + lane_hoff(lane);
    v8f acc[4] = {};
#pragma unroll
    for (int f = 0; f < 32; ++f) {
        int kt = f >> 2, t = f & 3;
        v16h b = lds_b_frag(wl + ((size_t)kt * 8 + t) * FRAG_H);
        acc[t] = __builtin_amdgcn_wmma_f32_16x16x32_f16(
            false, a[kt], false, b, (short)0, acc[t], false, false);
    }

    int nlane = lane & 15, mh = (lane >> 4) * 8;
#pragma unroll
    for (int t = 0; t < 4; ++t) {
        int col = (chalf * 4 + t) * 16 + nlane;
        float bc = b2[col];
#pragma unroll
        for (int j = 0; j < 8; ++j) {
            int row = rowbase + mh + j;
            if (row < N) {
                size_t o = (size_t)row * DD + col;
                out[o] = hres[o] + acc[t][j] + bc;
            }
        }
    }
}

extern "C" void kernel_launch(void* const* d_in, const int* in_sizes, int n_in,
                              void* d_out, int out_size, void* d_ws, size_t ws_size,
                              hipStream_t stream) {
    const float* x     = (const float*)d_in[0];
    const int*   src   = (const int*)d_in[1];
    const int*   dst   = (const int*)d_in[2];
    const int*   gids  = (const int*)d_in[3];
    const float* temb  = (const float*)d_in[4];
    const float* Wsrc  = (const float*)d_in[5];
    const float* bsrc  = (const float*)d_in[6];
    const float* Wdst  = (const float*)d_in[7];
    const float* bdst  = (const float*)d_in[8];
    const float* attn  = (const float*)d_in[9];
    const float* gbias = (const float*)d_in[10];
    const float* ln1g  = (const float*)d_in[11];
    const float* ln1b  = (const float*)d_in[12];
    const float* ln2g  = (const float*)d_in[13];
    const float* ln2b  = (const float*)d_in[14];
    const float* tW    = (const float*)d_in[15];
    const float* tb    = (const float*)d_in[16];
    const float* W1    = (const float*)d_in[17];
    const float* b1    = (const float*)d_in[18];
    const float* W2    = (const float*)d_in[19];
    const float* b2    = (const float*)d_in[20];

    int N  = in_sizes[0] / DD;
    int E  = in_sizes[1];
    int Bn = in_sizes[4] / 128;
    float* out = (float*)d_out;

    // Workspace carve-up (256B aligned slots, heavy buffers reused)
    char* ws = (char*)d_ws;
    size_t off = 0;
    auto alloc = [&](size_t bytes) -> void* {
        void* p = ws + off;
        off = (off + bytes + 255) & ~(size_t)255;
        return p;
    };
    _Float16* wsrcz = (_Float16*)alloc((size_t)32 * FRAG_H * 2);
    _Float16* wdstz = (_Float16*)alloc((size_t)32 * FRAG_H * 2);
    _Float16* w1z   = (_Float16*)alloc((size_t)64 * FRAG_H * 2);
    _Float16* w2z   = (_Float16*)alloc((size_t)64 * FRAG_H * 2);
    float*    tcond = (float*)alloc((size_t)Bn * DD * 4);
    float*    nmax  = (float*)alloc((size_t)N * HH * 4);
    float*    den   = (float*)alloc((size_t)N * HH * 4);
    _Float16* h16   = (_Float16*)alloc((size_t)N * DD * 2);  // later: h2in16
    float*    fsrc  = (float*)alloc((size_t)N * DD * 4);     // later: y16 [N,256] f16
    float*    fdst  = (float*)alloc((size_t)N * DD * 4);     // later: rst -> hres
    float*    scores = (float*)alloc((size_t)E * HH * 4);    // later: ex (in place)

    // --- phase 0: weight swizzles, time MLP, softmax-state init ---
    prep_weight_swz<<<(DD * DD + 255) / 256, 256, 0, stream>>>(Wsrc, wsrcz, DD, DD);
    prep_weight_swz<<<(DD * DD + 255) / 256, 256, 0, stream>>>(Wdst, wdstz, DD, DD);
    prep_weight_swz<<<(DD * 2 * DD + 255) / 256, 256, 0, stream>>>(W1, w1z, DD, 2 * DD);
    prep_weight_swz<<<(2 * DD * DD + 255) / 256, 256, 0, stream>>>(W2, w2z, 2 * DD, DD);
    time_cond_kernel<<<(Bn * DD + 255) / 256, 256, 0, stream>>>(temb, tW, tb, tcond, Bn);
    init_softmax_kernel<<<(N * HH + 255) / 256, 256, 0, stream>>>(nmax, den, N * HH);

    // --- phase 1: LN1 + dual projection GEMM (WMMA, LDS-staged weights) ---
    ln1_kernel<<<(N + 7) / 8, 256, 0, stream>>>(x, ln1g, ln1b, h16, N);
    gemm_feat_kernel<<<(N + 63) / 64, 256, 0, stream>>>(h16, wsrcz, wdstz, bsrc, bdst,
                                                        fsrc, fdst, N);

    // --- phase 2: edge attention (L2-resident gathers + atomics) ---
    int edgeWaveBlocks = (E * 32 + 255) / 256;
    edge_score_kernel<<<edgeWaveBlocks, 256, 0, stream>>>(src, dst, fsrc, fdst, attn,
                                                          scores, nmax, E);
    float* rst = fdst;  // fdst dead after edge_score
    zero_kernel<<<(N * DD + 255) / 256, 256, 0, stream>>>(rst, N * DD);
    edge_exp_kernel<<<(E * HH + 255) / 256, 256, 0, stream>>>(dst, scores, nmax, den, E);
    edge_agg_kernel<<<edgeWaveBlocks, 256, 0, stream>>>(src, dst, fsrc, scores, den, rst, E);

    // --- phase 3: gelu + time-cond + residual + LN2 (fused) ---
    _Float16* h2in16 = h16;  // h16 dead after gemm_feat
    node_mid_kernel<<<(N + 7) / 8, 256, 0, stream>>>(x, gids, tcond, gbias, ln2g, ln2b,
                                                     rst, h2in16, N);

    // --- phase 4: FFN (WMMA, LDS-staged weights) + final residual ---
    _Float16* y16 = (_Float16*)fsrc;  // fsrc dead after edge_agg
    ffn1_kernel<<<(N + 63) / 64, 256, 0, stream>>>(h2in16, w1z, b1, y16, N);
    ffn2_kernel<<<(N + 63) / 64, 256, 0, stream>>>(y16, w2z, b2, rst /*hres*/, out, N);
}
